// Model_64501818851521
// MI455X (gfx1250) — matrix operand order repaired and checked by reading, not compiled, not run on hardware
//
#include <hip/hip_runtime.h>

// ---------------------------------------------------------------------------
// Sequential tanh-RNN scan (B=256, T=784, H=512) + linear head + NLL loss
// for MI455X (gfx1250): bf16 WMMA (v_wmma_f32_16x16x32_bf16), persistent
// kernel with device-wide per-step barrier. W_hh slice staged in 320KB-WGP
// LDS (conflict-free padded layout); h double-buffer stays L2-resident.
// ---------------------------------------------------------------------------

typedef __attribute__((ext_vector_type(16))) __bf16        v16bf;
typedef __attribute__((ext_vector_type(8)))  float         v8f;
typedef __attribute__((ext_vector_type(4)))  unsigned int  u32x4;

#define BB   256
#define TT   784
#define HH   512
#define NCL  10
#define NWG  8      // 2 (M) x 4 (N) workgroups of 128x128 output tile
#define WPAD 520    // padded LDS row stride (elements): 1040B = 260 dwords,
                    // 260 mod 64 = 4 -> 16-lane b128 reads hit all 64 banks

union Frag { v16bf v; u32x4 q[2]; };

// --- prep: fp32 W_hh -> bf16 (row-major [H][H], W[n][k]) -------------------
__global__ void conv_w_kernel(const float* __restrict__ W,
                              __bf16* __restrict__ Wbf, int n) {
  int i = blockIdx.x * blockDim.x + threadIdx.x;
  if (i < n) Wbf[i] = (__bf16)W[i];
}

// --- prep: xs[t][b] = inputs[b][order[t]]  (fp32, [T][B]) ------------------
__global__ void prep_x_kernel(const float* __restrict__ inputs,
                              const int* __restrict__ order,
                              float* __restrict__ xs) {
  int i = blockIdx.x * blockDim.x + threadIdx.x;
  if (i < TT * BB) {
    int t = i / BB;
    int b = i - t * BB;
    xs[i] = inputs[(size_t)b * TT + order[t]];
  }
}

// --- prep: zero h0 (bf16) and the grid-barrier counter ---------------------
__global__ void init_kernel(unsigned* __restrict__ h0u,
                            unsigned* __restrict__ cnt) {
  int i = blockIdx.x * blockDim.x + threadIdx.x;
  if (i == 0) *cnt = 0u;
  if (i < (BB * HH) / 2) h0u[i] = 0u;   // two bf16 zeros per dword
}

// --- persistent RNN scan ---------------------------------------------------
// Grid: 8 WGs (mblk = wg>>2 in {0,1}, nblk = wg&3 in {0..3}), 256 thr = 8 waves.
// Wave tile: 64(M) x 32(N) -> 4x2 accumulators of 16x16 f32.
// A (h rows) fragments from global/L2; B (W cols) fragments from LDS.
__global__ __launch_bounds__(256, 1)
void rnn_scan_kernel(const float* __restrict__ xs,
                     const float* __restrict__ w_ih,
                     const float* __restrict__ b_h,
                     const __bf16* __restrict__ Wbf,
                     __bf16* __restrict__ hbuf0,
                     __bf16* __restrict__ hbuf1,
                     unsigned* __restrict__ cnt)
{
  extern __shared__ char smem[];
  __bf16* Wlds = (__bf16*)smem;                 // [128][WPAD] bf16

  const int wg   = blockIdx.x;
  const int mblk = wg >> 2;
  const int nblk = wg & 3;
  const int tid  = threadIdx.x;
  const int wave = tid >> 5;
  const int lane = tid & 31;
  const int lo   = lane & 15;
  const int hi   = lane >> 4;

  const int wm = mblk * 128 + (wave >> 2) * 64;  // wave row base (global)
  const int wn = nblk * 128 + (wave & 3) * 32;   // wave col base (global)

  // ---- one-time stage of this WG's 128-column W slice into LDS ----
  {
    const u32x4* src = reinterpret_cast<const u32x4*>(Wbf + (size_t)nblk * 128 * HH);
    for (int i = tid; i < 128 * (HH / 8); i += 256) {  // 64 x 16B chunks/row
      const int r = i >> 6;
      const int c = i & 63;
      u32x4* drow = reinterpret_cast<u32x4*>(Wlds + (size_t)r * WPAD);
      drow[c] = src[(size_t)r * 64 + c];
    }
    __syncthreads();
  }

  // per-lane epilogue constants (columns this lane owns)
  const int col0 = wn + lo;
  const int col1 = wn + 16 + lo;
  const float wih[2] = { w_ih[col0], w_ih[col1] };
  const float bhv[2] = { b_h[col0],  b_h[col1]  };

  // A-fragment base offsets: row = wm + mi*16 + lo, lane-half K offset hi*8
  size_t aoff[4];
#pragma unroll
  for (int mi = 0; mi < 4; ++mi)
    aoff[mi] = (size_t)(wm + mi * 16 + lo) * HH + hi * 8;
  // B-fragment LDS offsets: local col, padded stride, lane-half K offset hi*8
  const int colL0 = (wave & 3) * 32 + lo;
  size_t boffl[2];
  boffl[0] = (size_t)colL0 * WPAD + hi * 8;
  boffl[1] = (size_t)(colL0 + 16) * WPAD + hi * 8;

#pragma unroll 1
  for (int t = 0; t < TT; ++t) {
    const __bf16* __restrict__ hs = (t & 1) ? hbuf1 : hbuf0;
    __bf16* __restrict__ hd       = (t & 1) ? hbuf0 : hbuf1;

    v8f acc[4][2];
#pragma unroll
    for (int mi = 0; mi < 4; ++mi)
#pragma unroll
      for (int ni = 0; ni < 2; ++ni)
        acc[mi][ni] = (v8f){0.f, 0.f, 0.f, 0.f, 0.f, 0.f, 0.f, 0.f};

#pragma unroll 4
    for (int kc = 0; kc < HH; kc += 32) {
      Frag a[4], bf[2];
#pragma unroll
      for (int mi = 0; mi < 4; ++mi) {
        const u32x4* p = reinterpret_cast<const u32x4*>(hs + aoff[mi] + kc);
        a[mi].q[0] = p[0];  // K = kc + hi*8 .. +7
        a[mi].q[1] = p[2];  // K = kc + hi*8 + 16 .. +23
      }
#pragma unroll
      for (int ni = 0; ni < 2; ++ni) {
        const u32x4* p = reinterpret_cast<const u32x4*>(Wlds + boffl[ni] + kc);
        bf[ni].q[0] = p[0]; // K = kc + hi*8 .. +7   (ds_load_b128)
        bf[ni].q[1] = p[2]; // K = kc + hi*8 + 16 .. +23
      }
#pragma unroll
      for (int mi = 0; mi < 4; ++mi)
#pragma unroll
        for (int ni = 0; ni < 2; ++ni)
          acc[mi][ni] = __builtin_amdgcn_wmma_f32_16x16x32_bf16(
              false, a[mi].v, false, bf[ni].v,
              (short)0, acc[mi][ni], false, false);
    }

    // epilogue: + x_t*w_ih + b_h, tanh, store bf16 h_{t+1}
    const float* __restrict__ xst = xs + (size_t)t * BB;
    if (t + 1 < TT)
      __builtin_prefetch(xs + (size_t)(t + 1) * BB + wm, 0, 1);
#pragma unroll
    for (int mi = 0; mi < 4; ++mi) {
      const int rbase = wm + mi * 16 + hi * 8;   // C layout: M = vgpr + hi*8
      float xr[8];
#pragma unroll
      for (int e = 0; e < 8; ++e) xr[e] = xst[rbase + e];
#pragma unroll
      for (int ni = 0; ni < 2; ++ni) {
        const int col = ni ? col1 : col0;
#pragma unroll
        for (int e = 0; e < 8; ++e) {
          float v = acc[mi][ni][e] + xr[e] * wih[ni] + bhv[ni];
          hd[(size_t)(rbase + e) * HH + col] = (__bf16)tanhf(v);
        }
      }
    }

    // device-wide generation barrier across the 8 persistent WGs
    __threadfence();
    __syncthreads();
    if (tid == 0) {
      __hip_atomic_fetch_add(cnt, 1u, __ATOMIC_RELEASE, __HIP_MEMORY_SCOPE_AGENT);
      const unsigned tgt = (unsigned)(t + 1) * NWG;
      while (__hip_atomic_load(cnt, __ATOMIC_ACQUIRE, __HIP_MEMORY_SCOPE_AGENT) < tgt)
        __builtin_amdgcn_s_sleep(2);
    }
    __syncthreads();
    __threadfence();
  }
}

// --- head: logits = h @ lin_W^T + lin_b; log_softmax; loss + correct -------
__global__ __launch_bounds__(256, 1)
void head_kernel(const __bf16* __restrict__ hfin,
                 const float* __restrict__ linW,
                 const float* __restrict__ linb,
                 const int* __restrict__ y,
                 float* __restrict__ out)
{
  __shared__ float sl[256];
  __shared__ int   sc[256];
  const int b = threadIdx.x;

  float logits[NCL];
#pragma unroll
  for (int c = 0; c < NCL; ++c) logits[c] = linb[c];
  for (int k = 0; k < HH; ++k) {
    const float hv = (float)hfin[(size_t)b * HH + k];
#pragma unroll
    for (int c = 0; c < NCL; ++c) logits[c] += hv * linW[c * HH + k];
  }
  float m = logits[0];
#pragma unroll
  for (int c = 1; c < NCL; ++c) m = fmaxf(m, logits[c]);
  float s = 0.f;
#pragma unroll
  for (int c = 0; c < NCL; ++c) s += expf(logits[c] - m);
  const float lse = logf(s);
  const int yb = y[b];
  const float lossb = -(logits[yb] - m - lse);
  int pred = 0; float bv = logits[0];
#pragma unroll
  for (int c = 1; c < NCL; ++c)
    if (logits[c] > bv) { bv = logits[c]; pred = c; }  // first-max wins

  sl[b] = lossb;
  sc[b] = (pred == yb) ? 1 : 0;
  __syncthreads();
  for (int off = 128; off > 0; off >>= 1) {
    if (b < off) { sl[b] += sl[b + off]; sc[b] += sc[b + off]; }
    __syncthreads();
  }
  if (b == 0) {
    out[0] = sl[0] / (float)BB;   // mean NLL loss
    out[1] = (float)sc[0];        // correct count
  }
}

// ---------------------------------------------------------------------------
extern "C" void kernel_launch(void* const* d_in, const int* in_sizes, int n_in,
                              void* d_out, int out_size, void* d_ws, size_t ws_size,
                              hipStream_t stream)
{
  const float* inputs = (const float*)d_in[0];
  const int*   y      = (const int*)d_in[1];
  const int*   order  = (const int*)d_in[2];
  const float* w_ih   = (const float*)d_in[3];
  const float* W_hh   = (const float*)d_in[4];
  const float* b_h    = (const float*)d_in[5];
  const float* lin_W  = (const float*)d_in[6];
  const float* lin_b  = (const float*)d_in[7];
  float* out = (float*)d_out;

  // workspace layout (all L2-resident): counter | W bf16 | h0 | h1 | xs
  char* ws = (char*)d_ws;
  unsigned* cnt = (unsigned*)ws;                                   // 256 B pad
  __bf16* Wbf  = (__bf16*)(ws + 256);                              // 512 KB
  __bf16* h0   = (__bf16*)(ws + 256 + 524288);                     // 256 KB
  __bf16* h1   = (__bf16*)(ws + 256 + 524288 + 262144);            // 256 KB
  float*  xs   = (float*) (ws + 256 + 524288 + 2 * 262144);        // 784 KB

  conv_w_kernel<<<(HH * HH + 255) / 256, 256, 0, stream>>>(W_hh, Wbf, HH * HH);
  prep_x_kernel<<<(TT * BB + 255) / 256, 256, 0, stream>>>(inputs, order, xs);
  init_kernel<<<((BB * HH) / 2 + 255) / 256, 256, 0, stream>>>((unsigned*)h0, cnt);

  // dynamic LDS: 128 rows x WPAD bf16 = 133120 B (fits gfx1250's 320KB WGP LDS)
  const size_t lds_bytes = (size_t)128 * WPAD * sizeof(__bf16);
  rnn_scan_kernel<<<NWG, 256, lds_bytes, stream>>>(xs, w_ih, b_h, Wbf, h0, h1, cnt);
  // T = 784 is even -> final h lands back in h0
  head_kernel<<<1, 256, 0, stream>>>(h0, lin_W, lin_b, y, out);
}
